// DeformableAttention_39934605918297
// MI455X (gfx1250) — matrix operand order repaired and checked by reading, not compiled, not run on hardware
//
#include <hip/hip_runtime.h>
#include <hip/hip_bf16.h>

// ---------------------------------------------------------------------------
// Deformable attention for MI455X (gfx1250, wave32, WMMA).
// GEMMs use v_wmma_f32_16x16x32_bf16 (bf16 inputs, fp32 accumulate):
//  - LDS tiles laid out so each fragment loads as two ds_load_b128 per lane
//  - double-buffered staging overlaps HBM streaming with WMMA
//  - A-matrix (streamed once) loaded non-temporally so L2 keeps the reused
//    data (weights + `value` for the scattered bilinear gathers)
//  - edge (M % 128) handled by a separately-launched guarded instantiation,
//    so the hot path has no exec-mask manipulation.
// ---------------------------------------------------------------------------

typedef __attribute__((ext_vector_type(16))) __bf16        v16bf;
typedef __attribute__((ext_vector_type(8)))  float         v8f;
typedef __attribute__((ext_vector_type(4)))  float         v4f;
typedef __attribute__((ext_vector_type(4)))  unsigned int  v4u;
typedef __attribute__((ext_vector_type(2)))  unsigned int  v2u;

union Frag16 {
    v16bf v;
    unsigned short u[16];
    v4u q[2];
};

__device__ __forceinline__ unsigned short f32_to_bf16_rne(float f) {
    unsigned int u = __float_as_uint(f);
    u += 0x7FFFu + ((u >> 16) & 1u);   // round-to-nearest-even
    return (unsigned short)(u >> 16);
}
__device__ __forceinline__ unsigned int pack_bf16x2(float a, float b) {
    return ((unsigned int)f32_to_bf16_rne(b) << 16) | (unsigned int)f32_to_bf16_rne(a);
}

// Stage one K-step of tiles into LDS.
//   sA : [128][40]  row-major A tile (rows blockRow..+127, cols k0..k0+31)
//   sBt: [64][40]   TRANSPOSED B tile: sBt[col][k] (cols colBase..+63)
template <bool FULL>
__device__ __forceinline__ void stage_tiles(
    const float* __restrict__ A, const float* __restrict__ B,
    long blockRow, int colBase, int M, int k0, int tid,
    unsigned short (*sA)[40], unsigned short (*sBt)[40])
{
    // A: 128x32 as 4 x v4f per thread (global_load_b128 NT, coalesced),
    //    written as ds_store_b64 (4 bf16).  A is streamed exactly once.
#pragma unroll
    for (int j = 0; j < 4; ++j) {
        int idx4 = j * 256 + tid;          // 0..1023 float4 slots
        int r    = idx4 >> 3;              // row in tile (8 float4 per row)
        int c4   = idx4 & 7;
        long gr  = blockRow + r;
        v4f v = (v4f){0.f, 0.f, 0.f, 0.f};
        if (FULL || gr < (long)M)
            v = __builtin_nontemporal_load((const v4f*)&A[gr * 256 + k0 + c4 * 4]);
        v2u pk;
        pk.x = pack_bf16x2(v.x, v.y);
        pk.y = pack_bf16x2(v.z, v.w);
        *(v2u*)&sA[r][c4 * 4] = pk;
    }
    // B: 32x64 as 2 x v4f per thread (coalesced, L2-resident weights),
    //    transposed b16 stores.
#pragma unroll
    for (int j = 0; j < 2; ++j) {
        int idx4 = j * 256 + tid;          // 0..511 float4 slots
        int r    = idx4 >> 4;              // k within step (16 float4 per row)
        int c4   = idx4 & 15;
        v4f v = *(const v4f*)&B[(long)(k0 + r) * 256 + colBase + c4 * 4];
        sBt[c4 * 4 + 0][r] = f32_to_bf16_rne(v.x);
        sBt[c4 * 4 + 1][r] = f32_to_bf16_rne(v.y);
        sBt[c4 * 4 + 2][r] = f32_to_bf16_rne(v.z);
        sBt[c4 * 4 + 3][r] = f32_to_bf16_rne(v.w);
    }
}

// ---------------------------------------------------------------------------
// C[M,256] = A[M,256] @ B[256,256]   (fp32 in/out, bf16 WMMA compute)
// Block: 256 threads = 8 waves. Block tile: 128 rows x 64 cols.
// Wave w computes rows [w*16, w*16+16), all 4 column tiles (A-frag reuse x4).
// ---------------------------------------------------------------------------
template <bool FULL>
__global__ __launch_bounds__(256)
void gemm256_bf16(const float* __restrict__ A, const float* __restrict__ B,
                  float* __restrict__ C, int M, long rowOffset)
{
    __shared__ __align__(16) unsigned short sA [2][128][40];  // 2 x 10 KB
    __shared__ __align__(16) unsigned short sBt[2][64][40];   // 2 x  5 KB

    const int tid      = threadIdx.x;
    const int wave     = tid >> 5;
    const int lane     = tid & 31;
    const int laneHalf = lane >> 4;   // 0 or 1
    const int lane16   = lane & 15;
    const long blockRow = rowOffset + (long)blockIdx.x * 128;
    const int  colBase  = blockIdx.y * 64;

    v8f zero = {};
    v8f acc[4];
#pragma unroll
    for (int i = 0; i < 4; ++i) acc[i] = zero;

    stage_tiles<FULL>(A, B, blockRow, colBase, M, 0, tid, sA[0], sBt[0]);
    __syncthreads();

    int buf = 0;
    for (int k0 = 0; k0 < 256; k0 += 32) {
        // prefetch next tile into the other buffer (overlaps with WMMA below)
        if (k0 + 32 < 256)
            stage_tiles<FULL>(A, B, blockRow, colBase, M, k0 + 32, tid,
                              sA[buf ^ 1], sBt[buf ^ 1]);

        // A fragment (16x32 bf16), ISA layout:
        //   lanes 0-15 : M=lane,    K={0..7, 16..23}
        //   lanes16-31 : M=lane-16, K={8..15,24..31}
        // -> two contiguous 16B LDS loads per lane.
        Frag16 a;
        const int arow = wave * 16 + lane16;
        const int kh   = laneHalf * 8;
        a.q[0] = *(const v4u*)&sA[buf][arow][kh];
        a.q[1] = *(const v4u*)&sA[buf][arow][16 + kh];

        // B fragment (32x16 bf16): lane holds column lane&15,
        //   K = {0..7,16..23} (lanes 0-15) or {8..15,24..31} (lanes 16-31)
        // -> two 16B LDS loads in the transposed tile.
        const int kb = laneHalf * 8;
#pragma unroll
        for (int nt = 0; nt < 4; ++nt) {
            Frag16 b;
            b.q[0] = *(const v4u*)&sBt[buf][nt * 16 + lane16][kb];
            b.q[1] = *(const v4u*)&sBt[buf][nt * 16 + lane16][16 + kb];
            acc[nt] = __builtin_amdgcn_wmma_f32_16x16x32_bf16(
                false, a.v, false, b.v, (short)0, acc[nt], false, false);
        }
        __syncthreads();
        buf ^= 1;
    }

    // C/D layout: VGPR r, lane -> row = (lane>>4)*8 + r, col = lane&15
#pragma unroll
    for (int nt = 0; nt < 4; ++nt) {
#pragma unroll
        for (int r = 0; r < 8; ++r) {
            long gr = blockRow + wave * 16 + laneHalf * 8 + r;
            if (FULL || gr < (long)M)
                C[gr * 256 + colBase + nt * 16 + lane16] = acc[nt][r];
        }
    }
}

// ---------------------------------------------------------------------------
// Per-query projections: ref = sigmoid(q@Wref), off = q@Woff + bias,
// attn = softmax over L*P per head of q@Wattn.  Emits:
//   loc [row][h][l][p][2]  (normalized [0,1]-space sampling location)
//   attn[row][h][l][p]
// One block (256 thr) per query row.
// ---------------------------------------------------------------------------
__global__ __launch_bounds__(256)
void proj_kernel(const float* __restrict__ q,
                 const float* __restrict__ Wref,
                 const float* __restrict__ Woff,
                 const float* __restrict__ off_bias,
                 const float* __restrict__ Wattn,
                 float* __restrict__ loc,
                 float* __restrict__ attn)
{
    __shared__ float s_q[256];
    __shared__ float s_out[292];   // 0..191 off | 192..287 attn logits | 288..289 ref logits

    const long row = blockIdx.x;
    const int tid = threadIdx.x;
    s_q[tid] = q[row * 256 + tid];
    __syncthreads();

    for (int i = tid; i < 290; i += 256) {
        const float* w;
        int stride, c;
        float acc = 0.0f;
        if (i < 192)      { w = Woff;  stride = 192; c = i;       acc = off_bias[i]; }
        else if (i < 288) { w = Wattn; stride = 96;  c = i - 192; }
        else              { w = Wref;  stride = 2;   c = i - 288; }
#pragma unroll 8
        for (int k = 0; k < 256; ++k)
            acc = fmaf(s_q[k], w[k * stride + c], acc);
        s_out[i] = acc;
    }
    __syncthreads();

    // softmax over L*P = 12 per head (threads 0..7)
    if (tid < 8) {
        float m = -1e30f;
#pragma unroll
        for (int j = 0; j < 12; ++j) m = fmaxf(m, s_out[192 + tid * 12 + j]);
        float e[12], s = 0.0f;
#pragma unroll
        for (int j = 0; j < 12; ++j) { e[j] = __expf(s_out[192 + tid * 12 + j] - m); s += e[j]; }
        float inv = 1.0f / s;
#pragma unroll
        for (int j = 0; j < 12; ++j)
            attn[(row * 8 + tid) * 12 + j] = e[j] * inv;
    }
    // loc = sigmoid(ref_logit) + off   (layout matches Woff column order h,l,p,xy)
    if (tid < 192) {
        float r = 1.0f / (1.0f + __expf(-s_out[288 + (tid & 1)]));
        loc[row * 192 + tid] = r + s_out[tid];
    }
}

// ---------------------------------------------------------------------------
// Bilinear sampling + attention-weighted sum.
// One wave per (b, q, h); lane = channel d (HD=32).
// value: [B*Lv, 256] fp32 (col = h*32 + d), samp: [B*Lq, 256] fp32.
// gx = loc_x * w - 0.5 ; gy = loc_y * h - 0.5  (zeros padding outside).
// ---------------------------------------------------------------------------
__global__ __launch_bounds__(256)
void sample_kernel(const float* __restrict__ value,
                   const float* __restrict__ loc,
                   const float* __restrict__ attn,
                   float* __restrict__ samp)
{
    const int tid  = threadIdx.x;
    const int grp  = blockIdx.x * 8 + (tid >> 5);   // (row*8 + h)
    const int lane = tid & 31;
    const long row = grp >> 3;                      // b*2048 + q
    const int  h   = grp & 7;
    const long b   = row >> 11;                     // Lq = 2048

    const float* vbase = value + (b * 13125) * 256 + h * 32 + lane;
    const float* locp  = loc  + (row * 8 + h) * 24;
    const float* attp  = attn + (row * 8 + h) * 12;

    const int HS[3] = {100, 50, 25};
    const int ST[3] = {0, 10000, 12500};

    float acc = 0.0f;
#pragma unroll
    for (int l = 0; l < 3; ++l) {
        const int hw    = HS[l];            // square levels: h == w
        const int start = ST[l];
#pragma unroll
        for (int p = 0; p < 4; ++p) {
            const float lx = locp[(l * 4 + p) * 2 + 0];
            const float ly = locp[(l * 4 + p) * 2 + 1];
            const float wa = attp[l * 4 + p];
            const float gx = lx * (float)hw - 0.5f;
            const float gy = ly * (float)hw - 0.5f;
            const float x0f = floorf(gx), y0f = floorf(gy);
            const int x0 = (int)x0f, y0 = (int)y0f;
            const float wx = gx - x0f, wy = gy - y0f;
#pragma unroll
            for (int cy = 0; cy < 2; ++cy) {
#pragma unroll
                for (int cx = 0; cx < 2; ++cx) {
                    const int xi = x0 + cx, yi = y0 + cy;
                    const float wc = (cx ? wx : 1.0f - wx) * (cy ? wy : 1.0f - wy);
                    if (xi >= 0 && xi < hw && yi >= 0 && yi < hw) {   // wave-uniform
                        acc = fmaf(wa * wc,
                                   vbase[(long)(start + yi * hw + xi) * 256],
                                   acc);
                    }
                }
            }
        }
    }
    samp[row * 256 + h * 32 + lane] = acc;
}

// ---------------------------------------------------------------------------
// Launch
// ---------------------------------------------------------------------------
extern "C" void kernel_launch(void* const* d_in, const int* in_sizes, int n_in,
                              void* d_out, int out_size, void* d_ws, size_t ws_size,
                              hipStream_t stream)
{
    const float* x        = (const float*)d_in[0];   // [16,2048,256]
    const float* enc      = (const float*)d_in[1];   // [16,13125,256]
    const float* Wq       = (const float*)d_in[2];   // [256,256]
    const float* Wref     = (const float*)d_in[3];   // [256,2]
    const float* Woff     = (const float*)d_in[4];   // [256,192]
    const float* off_bias = (const float*)d_in[5];   // [192]
    const float* Wattn    = (const float*)d_in[6];   // [256,96]
    const float* Wv       = (const float*)d_in[7];   // [256,256]
    const float* Wout     = (const float*)d_in[8];   // [256,256]
    float* out = (float*)d_out;                      // [16,2048,256]

    const long MQ = 16L * 2048;     // 32768 query rows (divisible by 128)
    const long MV = 16L * 13125;    // 210000 value rows (1640 full + 1 partial)

    // workspace carve-up (~305 MB)
    char* ws = (char*)d_ws;
    float* q     = (float*)ws;                 ws += MQ * 256 * sizeof(float);
    float* value = (float*)ws;                 ws += MV * 256 * sizeof(float);
    float* loc   = (float*)ws;                 ws += MQ * 192 * sizeof(float);
    float* attn  = (float*)ws;                 ws += MQ * 96  * sizeof(float);
    float* samp  = (float*)ws;                 ws += MQ * 256 * sizeof(float);
    (void)ws_size; (void)in_sizes; (void)n_in; (void)out_size;

    dim3 blk(256);

    // 1) q = x @ Wq  (all blocks full)
    gemm256_bf16<true><<<dim3((unsigned)(MQ / 128), 4), blk, 0, stream>>>(
        x, Wq, q, (int)MQ, 0L);
    // 2) value = encoder_input @ Wv  (1640 full blocks + 1 guarded tail block)
    gemm256_bf16<true><<<dim3((unsigned)(MV / 128), 4), blk, 0, stream>>>(
        enc, Wv, value, (int)MV, 0L);
    gemm256_bf16<false><<<dim3(1, 4), blk, 0, stream>>>(
        enc, Wv, value, (int)MV, (MV / 128) * 128);
    // 3) per-query projections -> loc, attn
    proj_kernel<<<dim3((unsigned)MQ), blk, 0, stream>>>(q, Wref, Woff, off_bias, Wattn, loc, attn);
    // 4) bilinear sampling + attention-weighted sum -> samp
    sample_kernel<<<dim3((unsigned)MQ), blk, 0, stream>>>(value, loc, attn, samp);
    // 5) out = samp @ Wout  (all blocks full)
    gemm256_bf16<true><<<dim3((unsigned)(MQ / 128), 4), blk, 0, stream>>>(
        samp, Wout, out, (int)MQ, 0L);
}
